// BoxModel_46643344834650
// MI455X (gfx1250) — compile-verified, matrix operand order checked by reading, not run on hardware
//
#include <hip/hip_runtime.h>
#include <hip/hip_bf16.h>

#define VOCAB 32000
#define DIM   128
#define BATCH 64
#define NGRAM 4
#define GRP   8            // batch group held in WMMA accumulators per pass

typedef __attribute__((ext_vector_type(16))) _Float16 v16h;
typedef __attribute__((ext_vector_type(8)))  float    v8f;

#define F32_TINY 1.17549435e-38f
#define LOG2E    1.442695040888963f
#define LN2      0.693147180559945f

// Raw hardware transcendentals (v_exp_f32 = exp2, v_log_f32 = log2).
// Inputs to fast_log are always >= FLT_MIN (normal), so no denorm fixup needed.
__device__ __forceinline__ float fast_exp(float x) {
    return __builtin_amdgcn_exp2f(x * LOG2E);
}
__device__ __forceinline__ float fast_log(float x) {
    return __builtin_amdgcn_logf(x) * LN2;
}
// log1p(e) for e in [0,1]: poly for small e (avoids 1+e cancellation), hw log otherwise.
__device__ __forceinline__ float log1p_fast(float e) {
    float p = e * (1.0f - e * (0.5f - 0.33333333f * e));
    float l = fast_log(1.0f + e);
    return (e < 0.03f) ? p : l;
}
__device__ __forceinline__ float softplus1(float x) {   // beta = 1
    return fmaxf(x, 0.0f) + log1p_fast(fast_exp(-fabsf(x)));
}
__device__ __forceinline__ float softplus10(float x) {  // beta = 10
    float bx = 10.0f * x;
    return 0.1f * (fmaxf(bx, 0.0f) + log1p_fast(fast_exp(-fabsf(bx))));
}

// ---- Kernel 1: context boxes. czZ[b][d][2] = {cz, cZ} ----
__global__ __launch_bounds__(256) void ctx_kernel(const int* __restrict__ x,
                                                  const float* __restrict__ wb,
                                                  float* __restrict__ czZ) {
    int t = blockIdx.x * blockDim.x + threadIdx.x;
    if (t >= BATCH * DIM) return;
    int b = t / DIM, d = t % DIM;
    float s0 = 0.0f, s1 = 0.0f;
#pragma unroll
    for (int n = 0; n < NGRAM; ++n) {
        int id = x[b * NGRAM + n];
        const float* r = wb + (size_t)id * 2 * DIM;
        s0 += r[d];
        s1 += r[DIM + d];
    }
    s0 *= 0.25f; s1 *= 0.25f;
    czZ[2 * (b * DIM + d) + 0] = s0;
    czZ[2 * (b * DIM + d) + 1] = s0 + softplus10(s1);
}

// ---- Kernel 2: dec[b,v] via WMMA row-sum reduction. Wave = 16 vocab rows. ----
__global__ __launch_bounds__(256) void dec_kernel(const float* __restrict__ wb,
                                                  const float* __restrict__ czZ,
                                                  const float* __restrict__ bias,
                                                  float* __restrict__ out) {
    // Stage the 64 KB context-box table in LDS: shared by all 8 waves,
    // inner-loop reads become broadcast ds_load_b128.
    __shared__ float s_czZ[BATCH * DIM * 2];
    {
        const float4* src = (const float4*)czZ;
        float4* dst = (float4*)s_czZ;
#pragma unroll
        for (int i = threadIdx.x; i < BATCH * DIM * 2 / 4; i += 256)
            dst[i] = src[i];
    }
    __syncthreads();

    const int lane = threadIdx.x & 31;
    const int wave = threadIdx.x >> 5;           // 8 waves -> 128 vocab rows per block
    const int m    = lane & 15;                  // A-matrix row this lane owns
    const int hi   = lane >> 4;                  // lane group selects K halves
    const int v0   = blockIdx.x * 128 + wave * 16;
    const int v    = v0 + m;

    const float* rowz = wb + (size_t)v * 2 * DIM;   // z row of this vocab word
    const float* rowd = rowz + DIM;                 // delta row

    v16h onesB;
#pragma unroll
    for (int i = 0; i < 16; ++i) onesB[i] = (_Float16)1.0f;

    for (int bg = 0; bg < BATCH; bg += GRP) {
        v8f acc[GRP];
#pragma unroll
        for (int g = 0; g < GRP; ++g) acc[g] = (v8f){0,0,0,0,0,0,0,0};

        for (int chunk = 0; chunk < 4; ++chunk) {
            const int kbase = chunk * 32 + hi * 8;  // ISA 16-bit A layout: lanes>=16 shift K by 8
            float vz[16], vZ[16];
#pragma unroll
            for (int i = 0; i < 16; ++i) {
                int k = kbase + ((i < 8) ? i : i + 8);
                float z = rowz[k];
                vz[i] = z;
                vZ[i] = z + softplus10(rowd[k]);
            }
#pragma unroll
            for (int g = 0; g < GRP; ++g) {
                const int b = bg + g;
                const float* c = s_czZ + 2 * b * DIM;
                v16h A;
#pragma unroll
                for (int i = 0; i < 16; ++i) {
                    int k = kbase + ((i < 8) ? i : i + 8);
                    float cz = c[2 * k], cZ = c[2 * k + 1];
                    float mz = fmaxf(cz, vz[i]);
                    float mZ = fminf(cZ, vZ[i]);
                    float side = softplus1(mZ - mz);
                    A[i] = (_Float16)fast_log(side + F32_TINY);
                }
                // D = A x ones + C : every column of D holds the 16 row-sums over this K-chunk
                acc[g] = __builtin_amdgcn_wmma_f32_16x16x32_f16(
                    false, A, false, onesB, (short)0, acc[g], false, false);
            }
        }
        // Extract column N=0: lane 0 holds M=0..7 in c[0..7], lane 16 holds M=8..15.
#pragma unroll
        for (int g = 0; g < GRP; ++g) {
            const int b = bg + g;
            if (m == 0) {
#pragma unroll
                for (int j = 0; j < 8; ++j) {
                    int vv = v0 + hi * 8 + j;
                    out[(size_t)b * VOCAB + vv] = acc[g][j] + bias[vv];
                }
            }
        }
    }
}

// ---- Kernel 3: in-place log_softmax over vocab per batch row ----
__global__ __launch_bounds__(256) void lsm_kernel(float* __restrict__ out) {
    __shared__ float sred[256];
    const int b = blockIdx.x;
    float* row = out + (size_t)b * VOCAB;

    float mx = -3.402823466e38f;
    for (int i = threadIdx.x; i < VOCAB; i += 256) mx = fmaxf(mx, row[i]);
    sred[threadIdx.x] = mx; __syncthreads();
    for (int s = 128; s > 0; s >>= 1) {
        if (threadIdx.x < s) sred[threadIdx.x] = fmaxf(sred[threadIdx.x], sred[threadIdx.x + s]);
        __syncthreads();
    }
    mx = sred[0]; __syncthreads();

    float sum = 0.0f;
    for (int i = threadIdx.x; i < VOCAB; i += 256) sum += fast_exp(row[i] - mx);
    sred[threadIdx.x] = sum; __syncthreads();
    for (int s = 128; s > 0; s >>= 1) {
        if (threadIdx.x < s) sred[threadIdx.x] += sred[threadIdx.x + s];
        __syncthreads();
    }
    float lse = mx + fast_log(sred[0]);

    for (int i = threadIdx.x; i < VOCAB; i += 256) row[i] -= lse;
}

extern "C" void kernel_launch(void* const* d_in, const int* in_sizes, int n_in,
                              void* d_out, int out_size, void* d_ws, size_t ws_size,
                              hipStream_t stream) {
    const int*   x    = (const int*)d_in[0];
    const float* wb   = (const float*)d_in[1];
    const float* bias = (const float*)d_in[2];
    float* out = (float*)d_out;
    float* czZ = (float*)d_ws;                 // 64*128*2 floats = 64 KB

    ctx_kernel<<<(BATCH * DIM + 255) / 256, 256, 0, stream>>>(x, wb, czZ);
    dec_kernel<<<VOCAB / 128, 256, 0, stream>>>(wb, czZ, bias, out);
    lsm_kernel<<<BATCH, 256, 0, stream>>>(out);
}
